// ClassificationModel_53807350284495
// MI455X (gfx1250) — matrix operand order repaired and checked by reading, not compiled, not run on hardware
//
#include <hip/hip_runtime.h>

// ---------------------------------------------------------------------------
// CDNA5 / gfx1250 MeshGraphNets forward pass.
// All 128-wide matmuls run on v_wmma_f32_16x16x32_bf16 (wave32, 8 waves/WG,
// 16 rows per workgroup, one 16-col tile per wave). e/h kept fp32 in memory
// for residual accuracy; WMMA operands converted to bf16 on the fly.
// ---------------------------------------------------------------------------

typedef __attribute__((ext_vector_type(16))) __bf16 v16bf;
typedef __attribute__((ext_vector_type(8)))  float  v8f;

union FragB16 {
  uint4 q[2];
  v16bf v;
};

__device__ __forceinline__ unsigned short f2bf(float f) {
  unsigned int u = __float_as_uint(f);
  unsigned int r = u + 0x7FFFu + ((u >> 16) & 1u);   // round-to-nearest-even
  return (unsigned short)(r >> 16);
}

#define TSTRIDE_A 392   // bf16 elems; 784B rows, 16B aligned, dword stride 196
#define TSTRIDE_H 136   // bf16 elems; 272B rows
#define TSTRIDE_O 132   // fp32 elems

// One dense layer: act[16 x K] (LDS bf16) @ Wpacked[K x 128] (global bf16,
// pre-swizzled to B-fragment layout) -> 16x128. Wave w owns cols [16w,16w+16).
__device__ __forceinline__ void do_layer(
    const unsigned short* __restrict__ actIn, int strideIn, int K,
    const uint4* __restrict__ wp, const float* __restrict__ bias,
    unsigned short* __restrict__ actOut, float* __restrict__ outF, bool relu)
{
  const int lane = threadIdx.x & 31;
  const int wave = threadIdx.x >> 5;
  const int row  = lane & 15;
  const int hh   = lane >> 4;

  v8f c = {0.f, 0.f, 0.f, 0.f, 0.f, 0.f, 0.f, 0.f};
  const int nkt = K >> 5;
  for (int kt = 0; kt < nkt; ++kt) {
    FragB16 fa, fb;
    const unsigned short* ap = actIn + row * strideIn + kt * 32 + hh * 8;
    fa.q[0] = *(const uint4*)(ap);        // VGPR0-3: K = kb + hh*8 .. +8
    fa.q[1] = *(const uint4*)(ap + 16);   // VGPR4-7: K = kb+16 + hh*8 .. +8
    const uint4* bp = wp + ((size_t)(kt * 8 + wave) * 32 + lane) * 2;
    fb.q[0] = bp[0];
    fb.q[1] = bp[1];
    c = __builtin_amdgcn_wmma_f32_16x16x32_bf16(false, fa.v, false, fb.v,
                                                (short)0, c, false, false);
  }
  const int n  = wave * 16 + row;   // output column owned by this lane
  const float bv = bias[n];
  const int mh = hh * 8;            // C/D layout: VGPR j -> row j + 8*hh
#pragma unroll
  for (int j = 0; j < 8; ++j) {
    float v = c[j] + bv;
    if (relu) v = fmaxf(v, 0.f);
    if (actOut) actOut[(mh + j) * TSTRIDE_H + n] = f2bf(v);
    else        outF[(mh + j) * TSTRIDE_O + n]   = v;
  }
}

// mode 0: encoder   (in0 = raw features [M x K0], K0 zero-padded to K1pad)
// mode 1: edge MLP  (in0 = e, in1 = h, gather via eidx; residual into e;
//                    scatter-add new e into agg[dst])
// mode 2: node MLP  (in0 = h, in1 = agg; residual into h)
__global__ __launch_bounds__(256) void mlp4_kernel(
    int mode, const float* __restrict__ in0, const float* __restrict__ in1,
    const int* __restrict__ eidx, int E, int K0, int K1pad,
    const unsigned int* __restrict__ blob,
    float* __restrict__ outp, float* __restrict__ aggOut)
{
  __shared__ __align__(16) unsigned short actA[16 * TSTRIDE_A];
  __shared__ __align__(16) unsigned short actB[16 * TSTRIDE_H];
  __shared__ __align__(16) unsigned short actC[16 * TSTRIDE_H];
  __shared__ float resid[16 * 128];
  __shared__ float outF[16 * TSTRIDE_O];
  __shared__ float lnS[128], lnS2[128], lnMu[16], lnR[16];
  __shared__ int   idxS[32];

  const int tid = threadIdx.x;
  const int rowBase = blockIdx.x * 16;

  // ---- gather / stage input tile (bf16) ------------------------------------
  if (mode == 1) {
    if (tid < 16) {
      idxS[tid]      = eidx[rowBase + tid];       // src
      idxS[16 + tid] = eidx[E + rowBase + tid];   // dst
    }
    __syncthreads();
    for (int i = tid; i < 16 * 384; i += 256) {
      int r = i / 384, cc = i - r * 384;
      float v;
      if (cc < 128) {
        v = in0[(size_t)(rowBase + r) * 128 + cc];
        resid[r * 128 + cc] = v;
      } else if (cc < 256) {
        v = in1[(size_t)idxS[r] * 128 + (cc - 128)];
      } else {
        v = in1[(size_t)idxS[16 + r] * 128 + (cc - 256)];
      }
      actA[r * TSTRIDE_A + cc] = f2bf(v);
    }
  } else if (mode == 2) {
    for (int i = tid; i < 16 * 256; i += 256) {
      int r = i >> 8, cc = i & 255;
      float v;
      if (cc < 128) {
        v = in0[(size_t)(rowBase + r) * 128 + cc];
        resid[r * 128 + cc] = v;
      } else {
        v = in1[(size_t)(rowBase + r) * 128 + (cc - 128)];
      }
      actA[r * TSTRIDE_A + cc] = f2bf(v);
    }
  } else {
    for (int i = tid; i < 16 * 32; i += 256) {
      int r = i >> 5, cc = i & 31;
      float v = (cc < K0) ? in0[(size_t)(rowBase + r) * K0 + cc] : 0.f;
      actA[r * TSTRIDE_A + cc] = f2bf(v);
    }
  }
  __syncthreads();

  // ---- param blob: packed bf16 weights then fp32 biases / LN ---------------
  const uint4* w1 = (const uint4*)blob;
  const uint4* w2 = (const uint4*)(blob + (size_t)K1pad * 64);
  const uint4* w3 = w2 + 2048;
  const uint4* w4 = w3 + 2048;
  const float* fp = (const float*)(blob + (size_t)K1pad * 64 + 3 * 8192);
  const float *b1 = fp, *b2 = fp + 128, *b3 = fp + 256, *b4 = fp + 384;
  const float *ga = fp + 512, *be = fp + 640;

  do_layer(actA, TSTRIDE_A, K1pad, w1, b1, actB, nullptr, true);
  __syncthreads();
  do_layer(actB, TSTRIDE_H, 128, w2, b2, actC, nullptr, true);
  __syncthreads();
  do_layer(actC, TSTRIDE_H, 128, w3, b3, actB, nullptr, true);
  __syncthreads();
  do_layer(actB, TSTRIDE_H, 128, w4, b4, nullptr, outF, false);
  __syncthreads();

  // ---- LayerNorm over feature dim (per row) --------------------------------
  if (tid < 128) {
    int r = tid >> 3, p = tid & 7;
    float s = 0.f, s2 = 0.f;
#pragma unroll
    for (int cc = 0; cc < 16; ++cc) {
      float v = outF[r * TSTRIDE_O + p * 16 + cc];
      s += v; s2 += v * v;
    }
    lnS[r * 8 + p] = s; lnS2[r * 8 + p] = s2;
  }
  __syncthreads();
  if (tid < 16) {
    float s = 0.f, s2 = 0.f;
    for (int p = 0; p < 8; ++p) { s += lnS[tid * 8 + p]; s2 += lnS2[tid * 8 + p]; }
    float mu  = s * (1.f / 128.f);
    float var = s2 * (1.f / 128.f) - mu * mu;
    lnMu[tid] = mu;
    lnR[tid]  = 1.f / sqrtf(var + 1e-5f);
  }
  __syncthreads();

  // ---- apply LN, residual, write out, fused scatter-add --------------------
  for (int i = tid; i < 16 * 128; i += 256) {
    int r = i >> 7, cc = i & 127;
    float v = (outF[r * TSTRIDE_O + cc] - lnMu[r]) * lnR[r] * ga[cc] + be[cc];
    if (mode != 0) v += resid[r * 128 + cc];
    outp[(size_t)(rowBase + r) * 128 + cc] = v;
    if (mode == 1)
      atomicAdd(&aggOut[(size_t)idxS[16 + r] * 128 + cc], v);
  }
}

// Pack fp32 W[K x 128] into the bf16 B-fragment layout:
// dword idx = ((kt*8 + nt)*32 + lane)*8 + j holds rows {kb, kb+1} of col
// nt*16 + (lane&15), kb = kt*32 + (lane>>4)*8 + 2j + (j>=4 ? 8 : 0). Rows >= K zero-filled.
__global__ void pack_w_kernel(const float* __restrict__ W, int K,
                              unsigned int* __restrict__ dst, int total)
{
  int idx = blockIdx.x * 256 + threadIdx.x;
  if (idx >= total) return;
  int j    = idx & 7;
  int lane = (idx >> 3) & 31;
  int f    = idx >> 8;
  int kt = f >> 3, nt = f & 7;
  int col = nt * 16 + (lane & 15);
  int k0  = kt * 32 + (lane >> 4) * 8 + 2 * j + (j >= 4 ? 8 : 0);
  unsigned int lo = (k0     < K) ? (unsigned int)f2bf(W[(size_t)k0 * 128 + col])       : 0u;
  unsigned int hi = (k0 + 1 < K) ? (unsigned int)f2bf(W[(size_t)(k0 + 1) * 128 + col]) : 0u;
  dst[idx] = lo | (hi << 16);
}

__global__ void zero_kernel(float* __restrict__ p, long n)
{
  long i = (long)blockIdx.x * blockDim.x + threadIdx.x;
  long step = (long)gridDim.x * blockDim.x;
  for (; i < n; i += step) p[i] = 0.f;
}

__global__ void mean_kernel(const float* __restrict__ h, int n,
                            float* __restrict__ acc)
{
  int c   = threadIdx.x & 127;
  int sub = threadIdx.x >> 7;       // 0 or 1
  int rbeg = blockIdx.x * 512;
  int rend = rbeg + 512; if (rend > n) rend = n;
  float s = 0.f;
  for (int r = rbeg + sub; r < rend; r += 2)
    s += h[(size_t)r * 128 + c];
  atomicAdd(&acc[c], s);
}

__global__ __launch_bounds__(128) void decode_kernel(
    const float* __restrict__ msum, int n,
    const float* __restrict__ W0, const float* __restrict__ B0,
    const float* __restrict__ W1, const float* __restrict__ B1,
    const float* __restrict__ W2, const float* __restrict__ B2,
    const float* __restrict__ W3, const float* __restrict__ B3,
    float* __restrict__ out)
{
  __shared__ float a[128], b[128];
  int t = threadIdx.x;
  a[t] = msum[t] / (float)n;
  __syncthreads();
  float s = B0[t];
  for (int k = 0; k < 128; ++k) s += a[k] * W0[k * 128 + t];
  b[t] = fmaxf(s, 0.f);
  __syncthreads();
  s = B1[t];
  for (int k = 0; k < 128; ++k) s += b[k] * W1[k * 128 + t];
  a[t] = fmaxf(s, 0.f);
  __syncthreads();
  s = B2[t];
  for (int k = 0; k < 128; ++k) s += a[k] * W2[k * 128 + t];
  b[t] = fmaxf(s, 0.f);
  __syncthreads();
  if (t == 0) {
    float o = B3[0];
    for (int k = 0; k < 128; ++k) o += b[k] * W3[k];
    out[0] = o;
  }
}

// ---------------------------------------------------------------------------
// Host side
// ---------------------------------------------------------------------------

static void build_blob(void* const* d_in, int base, int K0, int Kpad,
                       unsigned int* blob, hipStream_t s)
{
  unsigned int* wdst = blob;
  const int Ks[4]    = {K0,   128, 128, 128};
  const int Kpads[4] = {Kpad, 128, 128, 128};
  for (int l = 0; l < 4; ++l) {
    const float* W = (const float*)d_in[base + 2 * l];
    int total = Kpads[l] * 64;
    pack_w_kernel<<<(total + 255) / 256, 256, 0, s>>>(W, Ks[l], wdst, total);
    wdst += total;
  }
  float* fdst = (float*)wdst;
  for (int l = 0; l < 4; ++l)
    hipMemcpyAsync(fdst + l * 128, d_in[base + 2 * l + 1], 128 * sizeof(float),
                   hipMemcpyDeviceToDevice, s);
  hipMemcpyAsync(fdst + 512, d_in[base + 8], 128 * sizeof(float),
                 hipMemcpyDeviceToDevice, s);
  hipMemcpyAsync(fdst + 640, d_in[base + 9], 128 * sizeof(float),
                 hipMemcpyDeviceToDevice, s);
}

// Input order (jax pytree, dict keys sorted):
//  0: x [N,12]   1: edge_attr [E,7]   2: edge_index [2,E] (int32)
//  3.. : params leaves, order = blocks (10 x {edge mlp, node mlp}), dec,
//        edge_enc, node_enc; each LN-MLP = W0,b0,W1,b1,W2,b2,W3,b3,g,beta
extern "C" void kernel_launch(void* const* d_in, const int* in_sizes, int n_in,
                              void* d_out, int out_size, void* d_ws, size_t ws_size,
                              hipStream_t stream)
{
  const float* x   = (const float*)d_in[0];
  const float* eat = (const float*)d_in[1];
  const int*   eix = (const int*)d_in[2];
  const int N = in_sizes[0] / 12;
  const int E = in_sizes[1] / 7;

  char* w = (char*)d_ws;
  auto alloc = [&](size_t b) -> char* {
    char* p = w; w += (b + 255) & ~(size_t)255; return p;
  };
  float* h    = (float*)alloc((size_t)N * 128 * sizeof(float));
  float* e    = (float*)alloc((size_t)E * 128 * sizeof(float));
  float* agg  = (float*)alloc((size_t)N * 128 * sizeof(float));
  float* mbuf = (float*)alloc(128 * sizeof(float));

  auto blobBytes = [](int Kpad) -> size_t {
    return ((size_t)Kpad * 64 + 3 * 8192 + 768) * 4;
  };
  unsigned int* blobNE = (unsigned int*)alloc(blobBytes(32));
  unsigned int* blobEE = (unsigned int*)alloc(blobBytes(32));
  unsigned int* blobEb[10];
  unsigned int* blobNb[10];
  for (int b = 0; b < 10; ++b) {
    blobEb[b] = (unsigned int*)alloc(blobBytes(384));
    blobNb[b] = (unsigned int*)alloc(blobBytes(256));
  }

  // Repack weights to bf16 fragment layout (every call; deterministic).
  build_blob(d_in, 221, 12, 32, blobNE, stream);   // node_enc
  build_blob(d_in, 211, 7,  32, blobEE, stream);   // edge_enc
  for (int b = 0; b < 10; ++b) {
    build_blob(d_in, 3  + b * 20, 384, 384, blobEb[b], stream);  // block edge
    build_blob(d_in, 13 + b * 20, 256, 256, blobNb[b], stream);  // block node
  }

  // Encoders
  mlp4_kernel<<<N / 16, 256, 0, stream>>>(0, x, nullptr, nullptr, 0, 12, 32,
                                          blobNE, h, nullptr);
  mlp4_kernel<<<E / 16, 256, 0, stream>>>(0, eat, nullptr, nullptr, 0, 7, 32,
                                          blobEE, e, nullptr);

  // Message-passing blocks
  for (int b = 0; b < 10; ++b) {
    zero_kernel<<<2048, 256, 0, stream>>>(agg, (long)N * 128);
    mlp4_kernel<<<E / 16, 256, 0, stream>>>(1, e, h, eix, E, 384, 384,
                                            blobEb[b], e, agg);
    mlp4_kernel<<<N / 16, 256, 0, stream>>>(2, h, agg, nullptr, 0, 256, 256,
                                            blobNb[b], h, nullptr);
  }

  // Decoder: mean over nodes, then 128->128->128->128->1 MLP
  zero_kernel<<<1, 256, 0, stream>>>(mbuf, 128);
  mean_kernel<<<(N + 511) / 512, 256, 0, stream>>>(h, N, mbuf);
  decode_kernel<<<1, 128, 0, stream>>>(mbuf, N,
      (const float*)d_in[203], (const float*)d_in[204],
      (const float*)d_in[205], (const float*)d_in[206],
      (const float*)d_in[207], (const float*)d_in[208],
      (const float*)d_in[209], (const float*)d_in[210],
      (float*)d_out);
}
